// WaterNetworkGNN_84035330113686
// MI455X (gfx1250) — matrix operand order repaired and checked by reading, not compiled, not run on hardware
//
#include <hip/hip_runtime.h>

#define NNODES 50000
#define NEDGES 600000
#define HIDDEN 128
#define NLAYERS 6
#define WPB 4            // waves per block (small kernels)
#define TPB (WPB * 32)
#define EWPB 8           // waves per block (edge kernel)
#define ETPB (EWPB * 32)

typedef _Float16 v16h __attribute__((ext_vector_type(16)));
typedef _Float16 v8h  __attribute__((ext_vector_type(8)));
typedef float    v8f  __attribute__((ext_vector_type(8)));

__device__ __forceinline__ v16h mk_afrag(v8h lo, v8h hi) {
  return __builtin_shufflevector(lo, hi, 0,1,2,3,4,5,6,7,8,9,10,11,12,13,14,15);
}

__device__ __forceinline__ v8f wmma_f16(v16h a, v16h b, v8f c) {
  return __builtin_amdgcn_wmma_f32_16x16x32_f16(false, a, false, b, (short)0, c, false, false);
}

// ---------------------------------------------------------------------------
// Pack a row-major f32 weight [K][128] into WMMA B-fragment order (f16):
// P[((kk*8+nt)*32+lane)*16 + e] = W[(kk*32 + 16*(e>>3) + 8*(lane>>4) + (e&7))*128 + nt*16 + (lane&15)]
// ---------------------------------------------------------------------------
__global__ void pack_b_kernel(const float* __restrict__ W, _Float16* __restrict__ P, int K) {
  int o = blockIdx.x * blockDim.x + threadIdx.x;
  if (o >= K * HIDDEN) return;
  int e    = o & 15;
  int lane = (o >> 4) & 31;
  int nt   = (o >> 9) & 7;
  int kk   = o >> 12;
  int krow = kk * 32 + (e >> 3) * 16 + (lane >> 4) * 8 + (e & 7);
  int col  = nt * 16 + (lane & 15);
  P[o] = (_Float16)W[krow * HIDDEN + col];
}

__global__ void deg_kernel(const int* __restrict__ dst, float* __restrict__ deg) {
  int i = blockIdx.x * blockDim.x + threadIdx.x;
  if (i < NEDGES) atomicAdd(&deg[dst[i]], 1.0f);
}

__global__ void invdeg_kernel(float* __restrict__ deg) {
  int i = blockIdx.x * blockDim.x + threadIdx.x;
  if (i < NNODES) { float d = deg[i]; deg[i] = (d > 0.5f) ? (1.0f / d) : 0.0f; }
}

// ---------------------------------------------------------------------------
// Encoder: h = relu(x @ W1 + b1) @ W2 + b2
// ---------------------------------------------------------------------------
__global__ void encoder_kernel(const float* __restrict__ x,
                               const float* __restrict__ W1, const float* __restrict__ b1,
                               const _Float16* __restrict__ W2p, const float* __restrict__ b2,
                               float* __restrict__ h, _Float16* __restrict__ h16, int ntiles) {
  __shared__ __align__(16) _Float16 T[WPB][16][136];
  int wave = threadIdx.x >> 5, lane = threadIdx.x & 31;
  int tile = blockIdx.x * WPB + wave;
  if (tile >= ntiles) return;
  int m = lane & 15, hh = lane >> 4;
  int node = tile * 16 + m;
  float x0 = x[node * 3 + 0], x1 = x[node * 3 + 1], x2 = x[node * 3 + 2];
  for (int c4 = 0; c4 < 16; ++c4) {
    int c = hh * 64 + c4 * 4;
    float4 w0 = *(const float4*)(W1 + c);
    float4 w1 = *(const float4*)(W1 + 128 + c);
    float4 w2 = *(const float4*)(W1 + 256 + c);
    float4 bb = *(const float4*)(b1 + c);
    T[wave][m][c + 0] = (_Float16)fmaxf(fmaf(x0, w0.x, fmaf(x1, w1.x, fmaf(x2, w2.x, bb.x))), 0.f);
    T[wave][m][c + 1] = (_Float16)fmaxf(fmaf(x0, w0.y, fmaf(x1, w1.y, fmaf(x2, w2.y, bb.y))), 0.f);
    T[wave][m][c + 2] = (_Float16)fmaxf(fmaf(x0, w0.z, fmaf(x1, w1.z, fmaf(x2, w2.z, bb.z))), 0.f);
    T[wave][m][c + 3] = (_Float16)fmaxf(fmaf(x0, w0.w, fmaf(x1, w1.w, fmaf(x2, w2.w, bb.w))), 0.f);
  }
  v16h Afr[4];
#pragma unroll
  for (int kk = 0; kk < 4; ++kk)
    Afr[kk] = mk_afrag(*(const v8h*)&T[wave][m][kk * 32 + 8 * hh],
                       *(const v8h*)&T[wave][m][kk * 32 + 16 + 8 * hh]);
#pragma unroll
  for (int nt = 0; nt < 8; ++nt) {
    const v16h* Wk = ((const v16h*)W2p) + nt * 32 + lane;
    v8f a = {};
#pragma unroll
    for (int kk = 0; kk < 4; ++kk) a = wmma_f16(Afr[kk], Wk[(size_t)kk * 256], a);
    float bv = b2[nt * 16 + m];
#pragma unroll
    for (int r = 0; r < 8; ++r) {
      int row = 8 * hh + r, col = nt * 16 + m;
      size_t idx = (size_t)(tile * 16 + row) * HIDDEN + col;
      float v = a[r] + bv;
      h[idx] = v;
      h16[idx] = (_Float16)v;
    }
  }
}

// ---------------------------------------------------------------------------
// Edge message MLP. 8 waves/block, 2 tiles (32 edges) per wave.
// Full packed W1+W2 (128 KB) staged in LDS once per block (16 edge-tiles
// amortize it), exploiting CDNA5's 320 KB/workgroup LDS.
// ---------------------------------------------------------------------------
__global__ void __launch_bounds__(ETPB)
edge_msg_kernel(const _Float16* __restrict__ h16,
                const float* __restrict__ ea,
                const float* __restrict__ eeW, const float* __restrict__ eeb,
                const int* __restrict__ srcI, const int* __restrict__ dstI,
                const _Float16* __restrict__ W1p, const float* __restrict__ b1,
                const _Float16* __restrict__ W2p, const float* __restrict__ b2,
                float* __restrict__ aggr, int npairs) {
  __shared__ __align__(16) _Float16 Wbuf[(384 + 128) * HIDDEN];  // 131072 B
  __shared__ __align__(16) _Float16 T[EWPB][32][136];            //  69632 B
  int tid = threadIdx.x;
  // ---- cooperative weight staging: W1 pack (6144 uint4) + W2 pack (2048) ----
  {
    const uint4* s1 = (const uint4*)W1p;
    const uint4* s2 = (const uint4*)W2p;
    uint4* dW = (uint4*)Wbuf;
#pragma unroll 4
    for (int i = tid; i < 6144; i += ETPB) dW[i] = s1[i];
#pragma unroll 2
    for (int i = tid; i < 2048; i += ETPB) dW[6144 + i] = s2[i];
  }
  __syncthreads();

  int wave = tid >> 5, lane = tid & 31;
  int pair = blockIdx.x * EWPB + wave;
  if (pair >= npairs) return;
  int m = lane & 15, hh = lane >> 4;
  int eb = pair * 32;

  int sN[2], dN[2];
#pragma unroll
  for (int tt = 0; tt < 2; ++tt) {
    int edge = eb + tt * 16 + m;
    sN[tt] = srcI[edge];
    dN[tt] = dstI[edge];
    // edge-attr encoder -> LDS (f16)
    float a0 = ea[edge * 4 + 0], a1 = ea[edge * 4 + 1];
    float a2 = ea[edge * 4 + 2], a3 = ea[edge * 4 + 3];
    for (int c4 = 0; c4 < 16; ++c4) {
      int c = hh * 64 + c4 * 4;
      float4 w0 = *(const float4*)(eeW + c);
      float4 w1 = *(const float4*)(eeW + 128 + c);
      float4 w2 = *(const float4*)(eeW + 256 + c);
      float4 w3 = *(const float4*)(eeW + 384 + c);
      float4 bb = *(const float4*)(eeb + c);
      _Float16* tr = &T[wave][tt * 16 + m][0];
      tr[c + 0] = (_Float16)fmaxf(fmaf(a0, w0.x, fmaf(a1, w1.x, fmaf(a2, w2.x, fmaf(a3, w3.x, bb.x)))), 0.f);
      tr[c + 1] = (_Float16)fmaxf(fmaf(a0, w0.y, fmaf(a1, w1.y, fmaf(a2, w2.y, fmaf(a3, w3.y, bb.y)))), 0.f);
      tr[c + 2] = (_Float16)fmaxf(fmaf(a0, w0.z, fmaf(a1, w1.z, fmaf(a2, w2.z, fmaf(a3, w3.z, bb.z)))), 0.f);
      tr[c + 3] = (_Float16)fmaxf(fmaf(a0, w0.w, fmaf(a1, w1.w, fmaf(a2, w2.w, fmaf(a3, w3.w, bb.w)))), 0.f);
    }
  }

  // ---- Preload ALL A fragments for GEMM1 (2 tiles x 12 k-steps) ----
  v16h Afr[2][12];
#pragma unroll
  for (int tt = 0; tt < 2; ++tt) {
    const _Float16* rowD = h16 + (size_t)dN[tt] * HIDDEN;
    const _Float16* rowS = h16 + (size_t)sN[tt] * HIDDEN;
#pragma unroll
    for (int kk = 0; kk < 4; ++kk) {
      int kb = kk * 32;
      Afr[tt][kk] = mk_afrag(*(const v8h*)(rowD + kb + 8 * hh),
                             *(const v8h*)(rowD + kb + 16 + 8 * hh));
      Afr[tt][4 + kk] = mk_afrag(*(const v8h*)(rowS + kb + 8 * hh),
                                 *(const v8h*)(rowS + kb + 16 + 8 * hh));
      Afr[tt][8 + kk] = mk_afrag(*(const v8h*)&T[wave][tt * 16 + m][kb + 8 * hh],
                                 *(const v8h*)&T[wave][tt * 16 + m][kb + 16 + 8 * hh]);
    }
  }

  // ---- GEMM1: B fragments streamed from LDS ----
  const v16h* W1l = (const v16h*)Wbuf;
  const v16h* W2l = (const v16h*)(Wbuf + 384 * HIDDEN);
#pragma unroll
  for (int nt = 0; nt < 8; ++nt) {
    const v16h* Wk = W1l + nt * 32 + lane;
    v8f a0 = {}, a1 = {};
#pragma unroll
    for (int kk = 0; kk < 12; ++kk) {
      v16h B = Wk[(size_t)kk * 256];
      a0 = wmma_f16(Afr[0][kk], B, a0);
      a1 = wmma_f16(Afr[1][kk], B, a1);
    }
    float bv = b1[nt * 16 + m];
    // relu(acc+bias) -> f16 mid tiles in LDS (reuse T; A frags already in regs)
#pragma unroll
    for (int r = 0; r < 8; ++r) {
      T[wave][8 * hh + r][nt * 16 + m]      = (_Float16)fmaxf(a0[r] + bv, 0.f);
      T[wave][16 + 8 * hh + r][nt * 16 + m] = (_Float16)fmaxf(a1[r] + bv, 0.f);
    }
  }

  // ---- Preload A fragments for GEMM2 (from mid tiles) ----
  v16h A2[2][4];
#pragma unroll
  for (int tt = 0; tt < 2; ++tt)
#pragma unroll
    for (int kk = 0; kk < 4; ++kk)
      A2[tt][kk] = mk_afrag(*(const v8h*)&T[wave][tt * 16 + m][kk * 32 + 8 * hh],
                            *(const v8h*)&T[wave][tt * 16 + m][kk * 32 + 16 + 8 * hh]);

  // destination rows for the scatter (hoisted)
  int dRow[2][8];
#pragma unroll
  for (int tt = 0; tt < 2; ++tt)
#pragma unroll
    for (int r = 0; r < 8; ++r) dRow[tt][r] = dstI[eb + tt * 16 + 8 * hh + r];

  // ---- GEMM2: nt-outer + immediate scatter-add ----
#pragma unroll
  for (int nt = 0; nt < 8; ++nt) {
    const v16h* Wk = W2l + nt * 32 + lane;
    v8f a0 = {}, a1 = {};
#pragma unroll
    for (int kk = 0; kk < 4; ++kk) {
      v16h B = Wk[(size_t)kk * 256];
      a0 = wmma_f16(A2[0][kk], B, a0);
      a1 = wmma_f16(A2[1][kk], B, a1);
    }
    float bv = b2[nt * 16 + m];
#pragma unroll
    for (int r = 0; r < 8; ++r) {
      atomicAdd(aggr + (size_t)dRow[0][r] * HIDDEN + nt * 16 + m, a0[r] + bv);
      atomicAdd(aggr + (size_t)dRow[1][r] * HIDDEN + nt * 16 + m, a1[r] + bv);
    }
  }
}

// ---------------------------------------------------------------------------
// Node update: upd = relu([h, aggr*inv_deg] @ updW + b); h = LN(h + upd)
// ---------------------------------------------------------------------------
__global__ void node_update_kernel(float* __restrict__ h, _Float16* __restrict__ h16,
                                   const float* __restrict__ aggr, const float* __restrict__ invd,
                                   const _Float16* __restrict__ Wp, const float* __restrict__ ub,
                                   const float* __restrict__ lng, const float* __restrict__ lnb,
                                   int ntiles) {
  __shared__ __align__(16) char NB[WPB][8448];  // 16x264 f16 A-tile, reused as 16x132 f32 x-buffer
  int wave = threadIdx.x >> 5, lane = threadIdx.x & 31;
  int tile = blockIdx.x * WPB + wave;
  if (tile >= ntiles) return;
  int m = lane & 15, hh = lane >> 4;
  _Float16 (*At)[264] = (_Float16 (*)[264]) & NB[wave][0];
  float    (*xb)[132] = (float (*)[132]) & NB[wave][0];
  int node = tile * 16 + m;
  // build A tile: cols 0..127 = h16[node], cols 128..255 = (f16)(aggr*inv_deg)
  if (hh == 0) {
    const v8h* srcp = (const v8h*)(h16 + (size_t)node * HIDDEN);
#pragma unroll
    for (int i = 0; i < 16; ++i) *(v8h*)&At[m][i * 8] = srcp[i];
  } else {
    float id = invd[node];
    const float* ar = aggr + (size_t)node * HIDDEN;
    for (int c = 0; c < 128; ++c) At[m][128 + c] = (_Float16)(ar[c] * id);
  }
  v16h Afr[8];
#pragma unroll
  for (int kk = 0; kk < 8; ++kk)
    Afr[kk] = mk_afrag(*(const v8h*)&At[m][kk * 32 + 8 * hh],
                       *(const v8h*)&At[m][kk * 32 + 16 + 8 * hh]);
#pragma unroll
  for (int nt = 0; nt < 8; ++nt) {
    const v16h* Wk = ((const v16h*)Wp) + nt * 32 + lane;
    v8f a = {};
#pragma unroll
    for (int kk = 0; kk < 8; ++kk) a = wmma_f16(Afr[kk], Wk[(size_t)kk * 256], a);
    float bv = ub[nt * 16 + m];
    // x = h + relu(upd) into f32 LDS buffer (A frags already in registers)
#pragma unroll
    for (int r = 0; r < 8; ++r) {
      int row = 8 * hh + r, col = nt * 16 + m;
      float hv = h[(size_t)(tile * 16 + row) * HIDDEN + col];
      xb[row][col] = hv + fmaxf(a[r] + bv, 0.f);
    }
  }
  // LayerNorm: one lane per row
  if (lane < 16) {
    int row = lane;
    float mu = 0.f;
    for (int c = 0; c < 128; ++c) mu += xb[row][c];
    mu *= (1.0f / 128.0f);
    float var = 0.f;
    for (int c = 0; c < 128; ++c) { float dl = xb[row][c] - mu; var += dl * dl; }
    var *= (1.0f / 128.0f);
    float rs = rsqrtf(var + 1e-5f);
    size_t base = (size_t)(tile * 16 + row) * HIDDEN;
    for (int c = 0; c < 128; ++c) {
      float y = (xb[row][c] - mu) * rs * lng[c] + lnb[c];
      h[base + c] = y;
      h16[base + c] = (_Float16)y;
    }
  }
}

// ---------------------------------------------------------------------------
// Decoder: pressure = relu(h @ W1 + b1) @ W2 + b2
// ---------------------------------------------------------------------------
__global__ void decoder_kernel(const _Float16* __restrict__ h16,
                               const _Float16* __restrict__ W1p, const float* __restrict__ b1,
                               const float* __restrict__ W2, const float* __restrict__ b2,
                               float* __restrict__ out, int ntiles) {
  __shared__ __align__(16) float T[WPB][16][132];
  int wave = threadIdx.x >> 5, lane = threadIdx.x & 31;
  int tile = blockIdx.x * WPB + wave;
  if (tile >= ntiles) return;
  int m = lane & 15, hh = lane >> 4;
  const _Float16* rowp = h16 + (size_t)(tile * 16 + m) * HIDDEN;
  v16h Afr[4];
#pragma unroll
  for (int kk = 0; kk < 4; ++kk)
    Afr[kk] = mk_afrag(*(const v8h*)(rowp + kk * 32 + 8 * hh),
                       *(const v8h*)(rowp + kk * 32 + 16 + 8 * hh));
#pragma unroll
  for (int nt = 0; nt < 8; ++nt) {
    const v16h* Wk = ((const v16h*)W1p) + nt * 32 + lane;
    v8f a = {};
#pragma unroll
    for (int kk = 0; kk < 4; ++kk) a = wmma_f16(Afr[kk], Wk[(size_t)kk * 256], a);
    float bv = b1[nt * 16 + m];
#pragma unroll
    for (int r = 0; r < 8; ++r)
      T[wave][8 * hh + r][nt * 16 + m] = fmaxf(a[r] + bv, 0.f);
  }
  if (lane < 16) {
    int row = lane;
    float p = b2[0];
    for (int c = 0; c < 128; ++c) p = fmaf(T[wave][row][c], W2[c], p);
    out[tile * 16 + row] = p;
  }
}

// ---------------------------------------------------------------------------
extern "C" void kernel_launch(void* const* d_in, const int* in_sizes, int n_in,
                              void* d_out, int out_size, void* d_ws, size_t ws_size,
                              hipStream_t stream) {
  (void)in_sizes; (void)n_in; (void)out_size; (void)ws_size;
  const float* x      = (const float*)d_in[0];
  const int*   ei     = (const int*)d_in[1];
  const float* ea     = (const float*)d_in[2];
  const float* encW1  = (const float*)d_in[3];
  const float* encb1  = (const float*)d_in[4];
  const float* encW2  = (const float*)d_in[5];
  const float* encb2  = (const float*)d_in[6];
  const float* eeW    = (const float*)d_in[7];
  const float* eeb    = (const float*)d_in[8];
  const float* msgW1  = (const float*)d_in[9];
  const float* msgb1  = (const float*)d_in[10];
  const float* msgW2  = (const float*)d_in[11];
  const float* msgb2  = (const float*)d_in[12];
  const float* updW   = (const float*)d_in[13];
  const float* updb   = (const float*)d_in[14];
  const float* lng    = (const float*)d_in[15];
  const float* lnb    = (const float*)d_in[16];
  const float* decW1  = (const float*)d_in[17];
  const float* decb1  = (const float*)d_in[18];
  const float* decW2  = (const float*)d_in[19];
  const float* decb2  = (const float*)d_in[20];
  float* out = (float*)d_out;
  const int* srcI = ei;
  const int* dstI = ei + NEDGES;

  char* w = (char*)d_ws;
  auto take = [&](size_t bytes) -> char* {
    char* p = w;
    w += (bytes + 255) & ~((size_t)255);
    return p;
  };
  float*    h      = (float*)take((size_t)NNODES * HIDDEN * 4);
  float*    aggr   = (float*)take((size_t)NNODES * HIDDEN * 4);
  _Float16* h16    = (_Float16*)take((size_t)NNODES * HIDDEN * 2);
  float*    invd   = (float*)take((size_t)NNODES * 4);
  _Float16* msgW1p = (_Float16*)take((size_t)NLAYERS * 384 * HIDDEN * 2);
  _Float16* msgW2p = (_Float16*)take((size_t)NLAYERS * HIDDEN * HIDDEN * 2);
  _Float16* updWp  = (_Float16*)take((size_t)NLAYERS * 256 * HIDDEN * 2);
  _Float16* encW2p = (_Float16*)take((size_t)HIDDEN * HIDDEN * 2);
  _Float16* decW1p = (_Float16*)take((size_t)HIDDEN * HIDDEN * 2);

  // ---- pack all GEMM weights into WMMA fragment order (f16) ----
  const int t128 = 128 * 128, t384 = 384 * 128, t256 = 256 * 128;
  pack_b_kernel<<<(t128 + 255) / 256, 256, 0, stream>>>(encW2, encW2p, 128);
  pack_b_kernel<<<(t128 + 255) / 256, 256, 0, stream>>>(decW1, decW1p, 128);
  for (int l = 0; l < NLAYERS; ++l) {
    pack_b_kernel<<<(t384 + 255) / 256, 256, 0, stream>>>(msgW1 + (size_t)l * t384, msgW1p + (size_t)l * t384, 384);
    pack_b_kernel<<<(t128 + 255) / 256, 256, 0, stream>>>(msgW2 + (size_t)l * t128, msgW2p + (size_t)l * t128, 128);
    pack_b_kernel<<<(t256 + 255) / 256, 256, 0, stream>>>(updW + (size_t)l * t256, updWp + (size_t)l * t256, 256);
  }

  // ---- degree / inverse degree ----
  (void)hipMemsetAsync(invd, 0, (size_t)NNODES * 4, stream);
  deg_kernel<<<(NEDGES + 255) / 256, 256, 0, stream>>>(dstI, invd);
  invdeg_kernel<<<(NNODES + 255) / 256, 256, 0, stream>>>(invd);

  const int ntilesN = NNODES / 16;       // 3125
  const int npairsE = NEDGES / 32;       // 18750 (two 16-edge tiles per wave)
  dim3 blkN(TPB), blkE(ETPB);
  dim3 gN((ntilesN + WPB - 1) / WPB);
  dim3 gE((npairsE + EWPB - 1) / EWPB);

  encoder_kernel<<<gN, blkN, 0, stream>>>(x, encW1, encb1, encW2p, encb2, h, h16, ntilesN);

  for (int l = 0; l < NLAYERS; ++l) {
    (void)hipMemsetAsync(aggr, 0, (size_t)NNODES * HIDDEN * 4, stream);
    edge_msg_kernel<<<gE, blkE, 0, stream>>>(h16, ea, eeW, eeb, srcI, dstI,
                                             msgW1p + (size_t)l * t384, msgb1 + l * HIDDEN,
                                             msgW2p + (size_t)l * t128, msgb2 + l * HIDDEN,
                                             aggr, npairsE);
    node_update_kernel<<<gN, blkN, 0, stream>>>(h, h16, aggr, invd,
                                                updWp + (size_t)l * t256, updb + l * HIDDEN,
                                                lng + l * HIDDEN, lnb + l * HIDDEN, ntilesN);
  }

  decoder_kernel<<<gN, blkN, 0, stream>>>(h16, decW1p, decb1, decW2, decb2, out, ntilesN);
}
